// FastSpeechModel_85684597555505
// MI455X (gfx1250) — matrix operand order repaired and checked by reading, not compiled, not run on hardware
//
#include <hip/hip_runtime.h>
#include <hip/hip_bf16.h>
#include <cstdint>
#include <cstddef>

// ---------------- types / constants ----------------
typedef __attribute__((ext_vector_type(16))) __bf16 v16bf;
typedef __attribute__((ext_vector_type(8)))  __bf16 v8bf;
typedef __attribute__((ext_vector_type(8)))  float  v8f;

#define Bc 16
#define Tc 256
#define MELc 1024
#define Dc 512
#define Hc 8
#define DHc 64
#define Lc 4
#define INTERc 2048
#define OUTc 80

__device__ __forceinline__ __bf16 f2bf(float f) {
  union { float f; uint32_t u; } c; c.f = f;
  uint32_t r = c.u + 0x7FFFu + ((c.u >> 16) & 1u);   // round-to-nearest-even
  uint16_t h = (uint16_t)(r >> 16);
  return __builtin_bit_cast(__bf16, h);
}
__device__ __forceinline__ float bf2f(__bf16 h) {
  uint32_t u = ((uint32_t)__builtin_bit_cast(uint16_t, h)) << 16;
  return __builtin_bit_cast(float, u);
}

// ---------------- weight cast f32[K][N] -> bf16[N][K] (per matrix) ----------------
__global__ void cast_tr_kernel(const float* __restrict__ src, __bf16* __restrict__ dst,
                               int Kd, int Nd, int mats) {
  size_t i = (size_t)blockIdx.x * blockDim.x + threadIdx.x;
  size_t total = (size_t)mats * Kd * Nd;
  if (i >= total) return;
  size_t mn = (size_t)Kd * Nd;
  size_t m = i / mn;
  size_t rem = i - m * mn;
  int k = (int)(rem / Nd);
  int n = (int)(rem - (size_t)k * Nd);
  dst[m * mn + (size_t)n * Kd + k] = f2bf(src[i]);
}

// ---------------- embedding + (reference's per-batch) positional enc ----------------
__global__ __launch_bounds__(256)
void embed_kernel(const int* __restrict__ tokens, const float* __restrict__ emb,
                  __bf16* __restrict__ x) {
  const int bt  = blockIdx.x;            // b*T + t
  const int b   = bt / Tc;
  const int tok = tokens[bt];
  #pragma unroll
  for (int j = 0; j < 2; ++j) {
    int d  = threadIdx.x + j * 256;
    int i2 = (d >> 1) * 2;
    float den = __expf(-(float)i2 * (9.210340371976184f / (float)Dc));
    float ang = (float)b * den;          // reference indexes pos by batch
    float p   = (d & 1) ? __cosf(ang) : __sinf(ang);
    x[(size_t)bt * Dc + d] = f2bf(2.f * emb[(size_t)tok * Dc + d] + p);
  }
}

// ---------------- duration cumsum + expansion (searchsorted right) ----------------
__global__ void cumsum_kernel(const int* __restrict__ dur, int* __restrict__ cums) {
  int b = threadIdx.x;
  if (b < Bc) {
    int s = 0;
    for (int j = 0; j < Tc; ++j) { s += dur[b * Tc + j]; cums[b * Tc + j] = s; }
  }
}

__global__ __launch_bounds__(256)
void expand_kernel(const __bf16* __restrict__ enc, const int* __restrict__ cums,
                   const int* __restrict__ mlen, __bf16* __restrict__ dec) {
  const int bf = blockIdx.x;             // b*MEL + f
  const int b = bf / MELc, fpos = bf % MELc;
  const int* c = cums + b * Tc;
  int lo = 0, hi = Tc;
  while (lo < hi) { int mid = (lo + hi) >> 1; if (c[mid] <= fpos) lo = mid + 1; else hi = mid; }
  int idx = lo < Tc ? lo : Tc - 1;
  const bool keep = (fpos <= mlen[b]);
  const __bf16 z = f2bf(0.f);
  #pragma unroll
  for (int j = 0; j < 2; ++j) {
    int d = threadIdx.x + j * 256;
    __bf16 v = enc[((size_t)b * Tc + idx) * Dc + d];
    dec[(size_t)bf * Dc + d] = keep ? v : z;
  }
}

// ---------------- fused residual + LayerNorm (bf16 in/out, f32 math) ----------------
__global__ __launch_bounds__(256)
void add_ln_kernel(const __bf16* __restrict__ x, const __bf16* __restrict__ t,
                   const float* __restrict__ gw, const float* __restrict__ bw,
                   __bf16* __restrict__ y) {
  __shared__ float sh[256];
  const size_t row = blockIdx.x;
  const int tid = threadIdx.x;
  const __bf16* xr = x + row * Dc;
  const __bf16* tr = t + row * Dc;
  float a0 = bf2f(xr[tid])       + bf2f(tr[tid]);
  float a1 = bf2f(xr[tid + 256]) + bf2f(tr[tid + 256]);
  sh[tid] = a0 + a1;
  __syncthreads();
  for (int st = 128; st > 0; st >>= 1) { if (tid < st) sh[tid] += sh[tid + st]; __syncthreads(); }
  float mean = sh[0] * (1.f / (float)Dc);
  __syncthreads();
  float d0 = a0 - mean, d1 = a1 - mean;
  sh[tid] = d0 * d0 + d1 * d1;
  __syncthreads();
  for (int st = 128; st > 0; st >>= 1) { if (tid < st) sh[tid] += sh[tid + st]; __syncthreads(); }
  float inv = rsqrtf(sh[0] * (1.f / (float)Dc) + 1e-5f);
  __syncthreads();
  __bf16* yr = y + row * Dc;
  yr[tid]       = f2bf(d0 * inv * gw[tid]       + bw[tid]);
  yr[tid + 256] = f2bf(d1 * inv * gw[tid + 256] + bw[tid + 256]);
}

// ---------------- WMMA GEMM: Y[M,N] = act( sum_tap X[m+shift] @ W[tap] + bias ) ----
// 256 threads = 8 waves (4M x 2N). WG tile 128x128, wave tile 32x64 (8 accums,
// 8 WMMA per K-step). Double-buffered LDS with register staging: one barrier per
// K-step; global loads for step+1 issue before the WMMAs of step.
// X bf16 [M][K]; W bf16 PRE-TRANSPOSED [tap][N][K]. A uses the k-permutation
// pc(k) = (k&7)+((k&16)>>1)+((k&8)<<1) so every fragment is one contiguous 32B
// LDS window (two ds_load_b128, no repacking).
template<int NTAPS, bool RELU, bool TRANS>
__global__ __launch_bounds__(256)
void gemm_kernel(const __bf16* __restrict__ X, const __bf16* __restrict__ W,
                 const float* __restrict__ bias, void* __restrict__ Yv,
                 int S, int Kdim, int Ndim) {
  __shared__ __bf16 As[2][128 * 32];     // [buf][row][pc(k)]  16 KB
  __shared__ __bf16 Bs[2][128 * 32];     // [buf][n][pc(k)]     16 KB
  const int tid  = threadIdx.x;
  const int lane = tid & 31, w = tid >> 5;
  const int wm = w & 3, wn = w >> 2;
  const int g  = lane >> 4, ln = lane & 15;
  const int m0 = blockIdx.y * 128;
  const int n0 = blockIdx.x * 128;

  const int srow = tid >> 2;             // staging row 0..63 (and +64)
  const int oct  = tid & 3;
  const int posA = ((oct & 1) << 4) | ((oct & 2) << 2);   // pc of octet base

  v8f acc[2][4] = {};
  const int KT = NTAPS * Kdim;
  const int nsteps = KT / 32;

  v8bf av[2], bvv[2];
  auto loadRegs = [&](int kk) {
    const int tap   = kk / Kdim;
    const int k0    = kk - tap * Kdim;
    const int shift = tap - NTAPS / 2;
    #pragma unroll
    for (int p = 0; p < 2; ++p) {
      int row = srow + 64 * p;
      int rg = m0 + row;
      int b = rg / S;
      int t = rg - b * S + shift;
      v8bf va = {};
      if (t >= 0 && t < S)
        va = *reinterpret_cast<const v8bf*>(X + ((size_t)b * S + t) * Kdim + k0 + oct * 8);
      av[p] = va;
      v8bf vb = {};
      if (n0 + row < Ndim)
        vb = *reinterpret_cast<const v8bf*>(
            W + (size_t)tap * Kdim * Ndim + (size_t)(n0 + row) * Kdim + k0 + oct * 8);
      bvv[p] = vb;
    }
  };
  auto storeLDS = [&](int buf) {
    #pragma unroll
    for (int p = 0; p < 2; ++p) {
      int row = srow + 64 * p;
      *reinterpret_cast<v8bf*>(&As[buf][row * 32 + posA])    = av[p];
      *reinterpret_cast<v8bf*>(&Bs[buf][row * 32 + posA])    = bvv[p];
    }
  };

  loadRegs(0);
  storeLDS(0);
  for (int step = 0; step < nsteps; ++step) {
    __syncthreads();                     // buf[cur] ready; prior reads complete
    const int cur = step & 1;
    const bool more = (step + 1) < nsteps;
    if (more) {
      loadRegs((step + 1) * 32);
      if (step + 2 < nsteps) {           // speculative prefetch of step+2 weights
        int kk2 = (step + 2) * 32;
        int tap2 = kk2 / Kdim, k2 = kk2 - tap2 * Kdim;
        __builtin_prefetch(W + (size_t)tap2 * Kdim * Ndim + (size_t)(n0 + srow) * Kdim + k2 + oct * 8, 0, 1);
      }
    }
    v16bf a[2], bb[4];
    #pragma unroll
    for (int i = 0; i < 2; ++i)
      a[i] = *reinterpret_cast<const v16bf*>(&As[cur][(wm * 32 + i * 16 + ln) * 32 + 16 * g]);
    #pragma unroll
    for (int j = 0; j < 4; ++j)
      bb[j] = *reinterpret_cast<const v16bf*>(&Bs[cur][(wn * 64 + j * 16 + ln) * 32 + 16 * g]);
    #pragma unroll
    for (int i = 0; i < 2; ++i)
      #pragma unroll
      for (int j = 0; j < 4; ++j)
        acc[i][j] = __builtin_amdgcn_wmma_f32_16x16x32_bf16(
            false, a[i], false, bb[j], (short)0, acc[i][j], false, false);
    if (more) storeLDS(cur ^ 1);
  }

  // ---- epilogue: bias (+ReLU); bf16 store, or f32 transposed store (final proj) ---
  #pragma unroll
  for (int i = 0; i < 2; ++i)
    #pragma unroll
    for (int j = 0; j < 4; ++j) {
      int col = n0 + wn * 64 + j * 16 + ln;
      if (col < Ndim) {
        float bvs = bias[col];
        #pragma unroll
        for (int r = 0; r < 8; ++r) {
          int row = m0 + wm * 32 + i * 16 + r + 8 * g;          // C/D: m = r + 8g
          float v = acc[i][j][r] + bvs;
          if (RELU) v = fmaxf(v, 0.f);
          if (TRANS) {
            int b2 = row / S, t2 = row - b2 * S;
            ((float*)Yv)[((size_t)b2 * Ndim + col) * S + t2] = v; // (B, OUT, MEL) f32
          } else {
            ((__bf16*)Yv)[(size_t)row * Ndim + col] = f2bf(v);
          }
        }
      }
    }
}

// ---------------- flash attention: block shares (b,h); wave = one 16-row Q tile ----
// K processed 64 columns per step: 8 score WMMAs + 8 ctx WMMAs per softmax round.
// qkv layout: [B, S, 3, H, DH] bf16. ctx: [B, S, H*DH] bf16.
__global__ __launch_bounds__(256)
void attn_kernel(const __bf16* __restrict__ qkv, const int* __restrict__ lens,
                 __bf16* __restrict__ ctx, int S) {
  __shared__ __bf16 Ks[64 * 64];         // [kr][d]        8 KB
  __shared__ __bf16 Vs[64 * 64];         // [d][kr]        8 KB
  __shared__ __bf16 Ps[8][16 * 64];      // per-wave P, [row][k], natural k order  16 KB
  const int tid  = threadIdx.x;
  const int w    = tid >> 5, lane = tid & 31;
  const int g    = lane >> 4, ln = lane & 15;
  const int b    = blockIdx.x / Hc, h = blockIdx.x % Hc;
  const int q0   = (blockIdx.y * 8 + w) * 16;
  const int RS   = 3 * Hc * DHc;         // 1536 bf16 per seq position
  const __bf16* qb = qkv + (size_t)b * S * RS + (size_t)h * DHc;
  const __bf16* kb = qb + Hc * DHc;
  const __bf16* vb = qb + 2 * Hc * DHc;
  const int len = lens[b];

  // Q A-fragments: two contiguous v8bf global loads each (16B aligned runs)
  v16bf aq[2];
  #pragma unroll
  for (int dc = 0; dc < 2; ++dc) {
    union { v16bf v; v8bf h8[2]; } u;
    const __bf16* qr = qb + (size_t)(q0 + ln) * RS + dc * 32;
    u.h8[0] = *reinterpret_cast<const v8bf*>(qr + 16 * g);       // k = 16g+e
    u.h8[1] = *reinterpret_cast<const v8bf*>(qr + 16 * g + 8);   // k = 16g+8+e
    aq[dc] = u.v;
  }

  float mrun[8], lrun[8];
  v8f acc[4] = {};
  #pragma unroll
  for (int r = 0; r < 8; ++r) { mrun[r] = -1e30f; lrun[r] = 0.f; }

  const int pl = ((ln & 8) << 1) | (ln & 7);   // pc(ln) within a 32-wide chunk

  for (int k0 = 0; k0 < S; k0 += 64) {
    // ---- block-cooperative K/V staging (64 rows x 64 d) ----
    #pragma unroll
    for (int p = 0; p < 2; ++p) {
      int idx8 = tid + 256 * p;
      int kr = idx8 >> 3, d8 = (idx8 & 7) * 8;
      v8bf kv = *reinterpret_cast<const v8bf*>(kb + (size_t)(k0 + kr) * RS + d8);
      *reinterpret_cast<v8bf*>(&Ks[kr * 64 + d8]) = kv;
      v8bf vv = *reinterpret_cast<const v8bf*>(vb + (size_t)(k0 + kr) * RS + d8);
      #pragma unroll
      for (int j = 0; j < 8; ++j) Vs[(d8 + j) * 64 + kr] = vv[j];
    }
    __syncthreads();

    // ---- scores: 4 x (16x16) tiles over K=DH=64 ----
    v8f sv[4] = {};
    #pragma unroll
    for (int dc = 0; dc < 2; ++dc)
      #pragma unroll
      for (int j = 0; j < 4; ++j) {
        v16bf bk = *reinterpret_cast<const v16bf*>(&Ks[(j * 16 + ln) * 64 + dc * 32 + 16 * g]);
        sv[j] = __builtin_amdgcn_wmma_f32_16x16x32_bf16(false, aq[dc], false, bk, (short)0, sv[j], false, false);
      }
    float msk[4];
    #pragma unroll
    for (int j = 0; j < 4; ++j) msk[j] = ((k0 + j * 16 + ln) > len) ? -1e30f : 0.f;

    #pragma unroll
    for (int r = 0; r < 8; ++r) {        // online softmax, rows m = r + 8g
      float vj[4];
      #pragma unroll
      for (int j = 0; j < 4; ++j) vj[j] = sv[j][r] * 0.125f + msk[j];   // 1/sqrt(DH)
      float tm = fmaxf(fmaxf(vj[0], vj[1]), fmaxf(vj[2], vj[3]));
      #pragma unroll
      for (int off = 1; off < 16; off <<= 1) tm = fmaxf(tm, __shfl_xor(tm, off, 32));
      float mnew = fmaxf(mrun[r], tm);
      float pj[4]; float rs = 0.f;
      #pragma unroll
      for (int j = 0; j < 4; ++j) { pj[j] = __expf(vj[j] - mnew); rs += pj[j]; }
      #pragma unroll
      for (int off = 1; off < 16; off <<= 1) rs += __shfl_xor(rs, off, 32);
      float sc = __expf(mrun[r] - mnew);
      lrun[r] = lrun[r] * sc + rs;
      mrun[r] = mnew;
      #pragma unroll
      for (int dt = 0; dt < 4; ++dt) acc[dt][r] *= sc;
      #pragma unroll
      for (int j = 0; j < 4; ++j)        // A-frag contiguous: chunk (j>>1), natural k within
        Ps[w][(r + 8 * g) * 64 + (j >> 1) * 32 + (j & 1) * 16 + ln] = f2bf(pj[j]);
    }
    // P as two A-fragments (k 0..31 and 32..63), contiguous 32B windows
    v16bf pa0 = *reinterpret_cast<const v16bf*>(&Ps[w][ln * 64 + 16 * g]);
    v16bf pa1 = *reinterpret_cast<const v16bf*>(&Ps[w][ln * 64 + 32 + 16 * g]);
    #pragma unroll
    for (int dt = 0; dt < 4; ++dt) {     // ctx(16x64) += P(16x64) @ V(64x64)
      v16bf bv0 = *reinterpret_cast<const v16bf*>(&Vs[(dt * 16 + ln) * 64 + 16 * g]);
      v16bf bv1 = *reinterpret_cast<const v16bf*>(&Vs[(dt * 16 + ln) * 64 + 32 + 16 * g]);
      acc[dt] = __builtin_amdgcn_wmma_f32_16x16x32_bf16(false, pa0, false, bv0, (short)0, acc[dt], false, false);
      acc[dt] = __builtin_amdgcn_wmma_f32_16x16x32_bf16(false, pa1, false, bv1, (short)0, acc[dt], false, false);
    }
    __syncthreads();                     // Ks/Vs reused next iteration
  }
  #pragma unroll
  for (int r = 0; r < 8; ++r) {
    float inv = 1.f / lrun[r];
    int q = q0 + r + 8 * g;
    #pragma unroll
    for (int dt = 0; dt < 4; ++dt)
      ctx[((size_t)b * S + q) * (Hc * DHc) + h * DHc + dt * 16 + ln] = f2bf(acc[dt][r] * inv);
  }
}

// ---------------- host orchestration ----------------
struct StackW {
  const __bf16 *wqkv, *wo, *c1w, *c2w;   // all pre-transposed [N][K] per tap
  const float *bqkv, *bo, *ln1g, *ln1b, *c1b, *c2b, *ln2g, *ln2b;
};

static void run_stack(__bf16* x, int S, const int* lens, const StackW& p,
                      __bf16* qkvb, __bf16* ctxb, __bf16* interb, __bf16* tmpb,
                      hipStream_t stream) {
  const int M = Bc * S;
  dim3 blk(256);
  for (int l = 0; l < Lc; ++l) {
    gemm_kernel<1, false, false><<<dim3((3 * Dc) / 128, M / 128), blk, 0, stream>>>(
        x, p.wqkv + (size_t)l * Dc * 3 * Dc, p.bqkv + l * 3 * Dc, qkvb, S, Dc, 3 * Dc);
    attn_kernel<<<dim3(Bc * Hc, S / 128), blk, 0, stream>>>(qkvb, lens, ctxb, S);
    gemm_kernel<1, false, false><<<dim3(Dc / 128, M / 128), blk, 0, stream>>>(
        ctxb, p.wo + (size_t)l * Dc * Dc, p.bo + l * Dc, tmpb, S, Dc, Dc);
    add_ln_kernel<<<M, blk, 0, stream>>>(x, tmpb, p.ln1g + l * Dc, p.ln1b + l * Dc, x);
    gemm_kernel<3, true, false><<<dim3(INTERc / 128, M / 128), blk, 0, stream>>>(
        x, p.c1w + (size_t)l * 3 * Dc * INTERc, p.c1b + l * INTERc, interb, S, Dc, INTERc);
    gemm_kernel<3, false, false><<<dim3(Dc / 128, M / 128), blk, 0, stream>>>(
        interb, p.c2w + (size_t)l * 3 * INTERc * Dc, p.c2b + l * Dc, tmpb, S, INTERc, Dc);
    add_ln_kernel<<<M, blk, 0, stream>>>(x, tmpb, p.ln2g + l * Dc, p.ln2b + l * Dc, x);
  }
}

extern "C" void kernel_launch(void* const* d_in, const int* in_sizes, int n_in,
                              void* d_out, int out_size, void* d_ws, size_t ws_size,
                              hipStream_t stream) {
  (void)in_sizes; (void)n_in; (void)out_size; (void)ws_size;
  const int*   tokens = (const int*)d_in[0];
  const int*   tlen   = (const int*)d_in[1];
  const int*   mlen   = (const int*)d_in[2];
  const int*   dur    = (const int*)d_in[3];
  const float* emb    = (const float*)d_in[5];
  const float* f[32];
  for (int i = 0; i < 32; ++i) f[i] = (const float*)d_in[i];

  char* ws = (char*)d_ws;
  size_t off = 0;
  auto alloc = [&](size_t bytes) -> void* {
    void* p = ws + off;
    off += (bytes + 255) & ~(size_t)255;
    return p;
  };

  const size_t n_qkv = (size_t)Lc * Dc * 3 * Dc;
  const size_t n_wo  = (size_t)Lc * Dc * Dc;
  const size_t n_c1  = (size_t)Lc * 3 * Dc * INTERc;
  const size_t n_c2  = (size_t)Lc * 3 * INTERc * Dc;
  const size_t n_ow  = (size_t)Dc * OUTc;

  __bf16* ewqkv = (__bf16*)alloc(n_qkv * 2);
  __bf16* ewo   = (__bf16*)alloc(n_wo * 2);
  __bf16* ec1   = (__bf16*)alloc(n_c1 * 2);
  __bf16* ec2   = (__bf16*)alloc(n_c2 * 2);
  __bf16* dwqkv = (__bf16*)alloc(n_qkv * 2);
  __bf16* dwo   = (__bf16*)alloc(n_wo * 2);
  __bf16* dc1   = (__bf16*)alloc(n_c1 * 2);
  __bf16* dc2   = (__bf16*)alloc(n_c2 * 2);
  __bf16* oww   = (__bf16*)alloc(n_ow * 2);

  const size_t NA = (size_t)Bc * MELc;   // max rows
  __bf16* xbuf   = (__bf16*)alloc(NA * Dc * 2);
  __bf16* x2buf  = (__bf16*)alloc(NA * Dc * 2);
  __bf16* tmpb   = (__bf16*)alloc(NA * Dc * 2);
  __bf16* ctxb   = (__bf16*)alloc(NA * Dc * 2);
  __bf16* qkvb   = (__bf16*)alloc(NA * 3 * Dc * 2);
  __bf16* interb = (__bf16*)alloc(NA * INTERc * 2);
  int*    cums   = (int*)alloc((size_t)Bc * Tc * 4);

  auto castT = [&](const float* s, __bf16* d, int Kd, int Nd, int mats) {
    size_t n = (size_t)mats * Kd * Nd;
    cast_tr_kernel<<<(unsigned)((n + 255) / 256), 256, 0, stream>>>(s, d, Kd, Nd, mats);
  };
  castT(f[6],  ewqkv, Dc, 3 * Dc, Lc);      castT(f[8],  ewo, Dc, Dc, Lc);
  castT(f[12], ec1,   Dc, INTERc, Lc * 3);  castT(f[14], ec2, INTERc, Dc, Lc * 3);
  castT(f[18], dwqkv, Dc, 3 * Dc, Lc);      castT(f[20], dwo, Dc, Dc, Lc);
  castT(f[24], dc1,   Dc, INTERc, Lc * 3);  castT(f[26], dc2, INTERc, Dc, Lc * 3);
  castT(f[30], oww,   Dc, OUTc, 1);

  // embedding + positional
  embed_kernel<<<Bc * Tc, 256, 0, stream>>>(tokens, emb, xbuf);

  // encoder
  StackW enc{ewqkv, ewo, ec1, ec2, f[7], f[9], f[10], f[11], f[13], f[15], f[16], f[17]};
  run_stack(xbuf, Tc, tlen, enc, qkvb, ctxb, interb, tmpb, stream);

  // duration expansion
  cumsum_kernel<<<1, 32, 0, stream>>>(dur, cums);
  expand_kernel<<<Bc * MELc, 256, 0, stream>>>(xbuf, cums, mlen, x2buf);

  // decoder
  StackW dec{dwqkv, dwo, dc1, dc2, f[19], f[21], f[22], f[23], f[25], f[27], f[28], f[29]};
  run_stack(x2buf, MELc, mlen, dec, qkvb, ctxb, interb, tmpb, stream);

  // output projection with fused (B, OUT, MEL) transpose (f32 output)
  gemm_kernel<1, false, true><<<dim3((OUTc + 127) / 128, (Bc * MELc) / 128), 256, 0, stream>>>(
      x2buf, oww, f[31], d_out, MELc, Dc, OUTc);
}
